// Net_34729105555716
// MI455X (gfx1250) — matrix operand order repaired and checked by reading, not compiled, not run on hardware
//
#include <hip/hip_runtime.h>

// ---------------- CDNA5 (gfx1250) fused RNN: embed + input-proj + 200-step
// ---------------- recurrence + fc1/relu + fc2, all bf16 WMMA w/ f32 accum.

typedef __attribute__((ext_vector_type(16))) __bf16        v16bf;
typedef __attribute__((ext_vector_type(8)))  float         v8f;
typedef __attribute__((ext_vector_type(8)))  unsigned int  v8u;

#define SEQ    200
#define BATCH  4096
#define EMB_R  100
#define EMB_P  128     // padded K for embedding (4 k-tiles of 32)
#define HID_R  300
#define HID_P  320     // padded K for hidden (10 k-tiles of 32)
#define NT_H   19      // N tiles covering 300 (19*16 = 304)
#define KT_H   10      // K tiles covering 300
#define KT_E   4       // K tiles covering 100
#define NT_F   16      // fc1 N tiles (256)

// ws layout (bytes)
#define WS_EMB    0
#define WS_BHH    7680000                // 30000*128*2
#define WS_BIH    (WS_BHH + 194560)      // 10*19*512*2
#define WS_BFC1   (WS_BIH + 77824)       // 4*19*512*2
#define WS_BIAS   (WS_BFC1 + 163840)     // 10*16*512*2

#define LDS_BIH_OFF   194560
#define LDS_STAGE_B   272384             // Bhh + Bih bytes staged
#define LDS_SCR_OFF   272384
#define LDS_TOTAL     313344             // + 4 waves * 10240B scratch

__device__ __forceinline__ unsigned short f2bf(float f) {
    unsigned int u = __builtin_bit_cast(unsigned int, f);
    unsigned int r = u + 0x7FFFu + ((u >> 16) & 1u);   // RNE
    return (unsigned short)(r >> 16);
}
__device__ __forceinline__ float bf2f(unsigned short b) {
    return __builtin_bit_cast(float, ((unsigned int)b) << 16);
}

// Hardware tanh on gfx1250 if the builtin exists; cheap exp2/rcp fallback.
__device__ __forceinline__ float tanh_fast(float x) {
#if __has_builtin(__builtin_amdgcn_tanhf)
    return __builtin_amdgcn_tanhf(x);
#elif __has_builtin(__builtin_amdgcn_tanh_f32)
    return __builtin_amdgcn_tanh_f32(x);
#else
    float e = __expf(2.0f * x);
#if __has_builtin(__builtin_amdgcn_rcpf)
    float r = __builtin_amdgcn_rcpf(e + 1.0f);
#else
    float r = 1.0f / (e + 1.0f);
#endif
    return 1.0f - 2.0f * r;
#endif
}

__device__ __forceinline__ unsigned short cvt_bf16(float f) {
    __bf16 h = (__bf16)f;
    return __builtin_bit_cast(unsigned short, h);
}

union FragU {
    uint4 q[2];
    v16bf v;
    unsigned short s[16];
};

// -------- pack kernels --------------------------------------------------

__global__ void pack_emb_kernel(const float* __restrict__ emb,
                                unsigned short* __restrict__ embP, int total) {
    int i = blockIdx.x * 256 + threadIdx.x;
    if (i >= total) return;
    int r = i >> 7, k = i & 127;
    embP[i] = (k < EMB_R) ? f2bf(emb[r * EMB_R + k]) : (unsigned short)0;
}

// Pack W [Nreal, Kreal] (row-major, used as out = in @ W^T) into B-fragment
// layout: frag (kt,nt), lane L holds n = nt*16 + (L&15), k = kt*32 + 16*(L>>4) + e,
// stored contiguously at P[frag*512 + L*16 + e].
__global__ void pack_frag_kernel(const float* __restrict__ W,
                                 unsigned short* __restrict__ P,
                                 int Nreal, int Kreal, int NTt, int total) {
    int i = blockIdx.x * 256 + threadIdx.x;
    if (i >= total) return;
    int f = i >> 9;
    int w = i & 511;
    int L = w >> 4, e = w & 15;
    int kt = f / NTt, nt = f - kt * NTt;
    int n = nt * 16 + (L & 15);
    int k = kt * 32 + (L >> 4) * 16 + e;
    float v = (n < Nreal && k < Kreal) ? W[n * Kreal + k] : 0.0f;
    P[i] = f2bf(v);
}

__global__ void pack_bias_kernel(const float* __restrict__ b_ih,
                                 const float* __restrict__ b_hh,
                                 float* __restrict__ biasP) {
    int i = threadIdx.x;  // 320 threads
    biasP[i] = (i < HID_R) ? (b_ih[i] + b_hh[i]) : 0.0f;
}

// -------- per-step pass (NTC accumulator tiles, to cap register peak) ----

template <int NT0, int NTC>
__device__ __forceinline__ void step_pass(
    const unsigned short* __restrict__ erow,
    const v8u* __restrict__ ldsBih,
    const v8u* __restrict__ ldsBhh,
    const v16bf (&hA)[KT_H],
    const float (&biasv)[NT_H],
    unsigned short* __restrict__ scr,
    int lane, int half, int nlo)
{
    v8f acc[NTC];
#pragma unroll
    for (int nt = 0; nt < NTC; ++nt) {
        v8f a;
#pragma unroll
        for (int j = 0; j < 8; ++j) a[j] = biasv[NT0 + nt];
        acc[nt] = a;
    }

    // xp contribution: xe @ w_ih^T
#pragma unroll
    for (int kt = 0; kt < KT_E; ++kt) {
        FragU a;
        const unsigned short* p = erow + kt * 32 + half * 16;
        a.q[0] = *(const uint4*)(p);
        a.q[1] = *(const uint4*)(p + 8);
#pragma unroll
        for (int nt = 0; nt < NTC; ++nt) {
            v16bf b = __builtin_bit_cast(
                v16bf, ldsBih[(kt * NT_H + NT0 + nt) * 32 + lane]);
            acc[nt] = __builtin_amdgcn_wmma_f32_16x16x32_bf16(
                false, a.v, false, b, (short)0, acc[nt], false, false);
        }
    }

    // recurrence: h @ w_hh^T
#pragma unroll
    for (int kt = 0; kt < KT_H; ++kt) {
#pragma unroll
        for (int nt = 0; nt < NTC; ++nt) {
            v16bf b = __builtin_bit_cast(
                v16bf, ldsBhh[(kt * NT_H + NT0 + nt) * 32 + lane]);
            acc[nt] = __builtin_amdgcn_wmma_f32_16x16x32_bf16(
                false, hA[kt], false, b, (short)0, acc[nt], false, false);
        }
    }

    // h = tanh(acc): C/D layout -> row-major bf16 tile in wave scratch.
#pragma unroll
    for (int nt = 0; nt < NTC; ++nt) {
#pragma unroll
        for (int j = 0; j < 8; ++j) {
            int m = half * 8 + j;
            scr[m * HID_P + (NT0 + nt) * 16 + nlo] =
                cvt_bf16(tanh_fast(acc[nt][j]));
        }
    }
}

// -------- fc1 pass -------------------------------------------------------

template <int NT0, int NTC>
__device__ __forceinline__ void fc1_pass(
    const v8u* __restrict__ gfc1,
    const v16bf (&hA)[KT_H],
    const float* __restrict__ fc1_b,
    unsigned short* __restrict__ scr,
    int lane, int half, int nlo)
{
    v8f acc[NTC];
#pragma unroll
    for (int nt = 0; nt < NTC; ++nt) {
        float bv = fc1_b[(NT0 + nt) * 16 + nlo];
        v8f a;
#pragma unroll
        for (int j = 0; j < 8; ++j) a[j] = bv;
        acc[nt] = a;
    }
#pragma unroll
    for (int kt = 0; kt < KT_H; ++kt) {
#pragma unroll
        for (int nt = 0; nt < NTC; ++nt) {
            v16bf b = __builtin_bit_cast(
                v16bf, gfc1[(kt * NT_F + NT0 + nt) * 32 + lane]);
            acc[nt] = __builtin_amdgcn_wmma_f32_16x16x32_bf16(
                false, hA[kt], false, b, (short)0, acc[nt], false, false);
        }
    }
    // relu, stash 16x256 bf16 tile (pitch 256)
#pragma unroll
    for (int nt = 0; nt < NTC; ++nt) {
#pragma unroll
        for (int j = 0; j < 8; ++j) {
            int m = half * 8 + j;
            float v = acc[nt][j];
            scr[m * 256 + (NT0 + nt) * 16 + nlo] =
                cvt_bf16(v > 0.0f ? v : 0.0f);
        }
    }
}

// -------- fused persistent RNN kernel -----------------------------------

__global__ __launch_bounds__(128) void rnn_fused_kernel(
    const int* __restrict__ x,
    const unsigned short* __restrict__ embP,
    const unsigned short* __restrict__ bfrags,   // BhhP followed by BihP
    const unsigned short* __restrict__ bfc1,     // fc1 fragments (stay in L2)
    const float* __restrict__ biasP,             // b_ih + b_hh, padded to 320
    const float* __restrict__ fc1_b,
    const float* __restrict__ fc2_w,
    const float* __restrict__ fc2_b,
    float* __restrict__ out)
{
    __shared__ uint4 smem4[LDS_TOTAL / 16];

    // Stage recurrent + input-proj weight fragments into LDS once.
    {
        const uint4* src = (const uint4*)bfrags;
        for (int i = threadIdx.x; i < LDS_STAGE_B / 16; i += 128)
            smem4[i] = src[i];
    }
    __syncthreads();

    const v8u* ldsBhh = (const v8u*)smem4;
    const v8u* ldsBih = (const v8u*)((const char*)smem4 + LDS_BIH_OFF);

    int t    = threadIdx.x;
    int wave = t >> 5;
    int lane = t & 31;
    int half = lane >> 4;
    int nlo  = lane & 15;
    unsigned short* scr =
        (unsigned short*)((char*)smem4 + LDS_SCR_OFF + wave * 10240);
    int b0 = blockIdx.x * 64 + wave * 16;

    // Zero the K-pad columns [300,320) of the wave-private h scratch.
    if (lane < 16)
        for (int k = HID_R; k < HID_P; ++k) scr[lane * HID_P + k] = 0;

    // Per-lane bias for accumulator init (N = nt*16 + nlo).
    float biasv[NT_H];
#pragma unroll
    for (int nt = 0; nt < NT_H; ++nt) biasv[nt] = biasP[nt * 16 + nlo];

    // h = 0, in A-fragment form.
    v16bf hA[KT_H];
    {
        FragU z;
#pragma unroll
        for (int i = 0; i < 16; ++i) z.s[i] = 0;
#pragma unroll
        for (int kt = 0; kt < KT_H; ++kt) hA[kt] = z.v;
    }

    for (int s = 0; s < SEQ; ++s) {
        // Embedding gather row for this step; prefetch next step's rows.
        int rowIdx = x[s * BATCH + b0 + nlo];
        const unsigned short* erow = embP + (size_t)rowIdx * EMB_P;
        if (s + 1 < SEQ) {
            int r2 = x[(s + 1) * BATCH + b0 + nlo];
            const char* pf = (const char*)(embP + (size_t)r2 * EMB_P);
            __builtin_prefetch(pf, 0, 0);
            __builtin_prefetch(pf + 128, 0, 0);
        }

        // Two accumulator passes (10 + 9 N-tiles) to stay under 256 VGPRs.
        step_pass<0, 10>(erow, ldsBih, ldsBhh, hA, biasv, scr, lane, half, nlo);
        step_pass<10, 9>(erow, ldsBih, ldsBhh, hA, biasv, scr, lane, half, nlo);

        // Reload h as A-fragments (wave-private, same-wave DS ordering).
#pragma unroll
        for (int kt = 0; kt < KT_H; ++kt) {
            FragU a;
            const unsigned short* p = scr + nlo * HID_P + kt * 32 + half * 16;
            a.q[0] = *(const uint4*)(p);
            a.q[1] = *(const uint4*)(p + 8);
            hA[kt] = a.v;
        }
    }

    // ----- fc1: relu(h @ fc1_w^T + fc1_b), B fragments streamed from L2 ---
    const v8u* gfc1 = (const v8u*)bfc1;
    fc1_pass<0, 8>(gfc1, hA, fc1_b, scr, lane, half, nlo);
    fc1_pass<8, 8>(gfc1, hA, fc1_b, scr, lane, half, nlo);

    // ----- fc2 (256 -> 3): one batch row per lane (lanes 0..15) -----------
    if (lane < 16) {
        int r = nlo;
        float s0 = fc2_b[0], s1 = fc2_b[1], s2 = fc2_b[2];
        for (int j = 0; j < 256; ++j) {
            float hv = bf2f(scr[r * 256 + j]);
            s0 += hv * fc2_w[j];
            s1 += hv * fc2_w[256 + j];
            s2 += hv * fc2_w[512 + j];
        }
        float* o = out + (size_t)(b0 + r) * 3;
        o[0] = s0; o[1] = s1; o[2] = s2;
    }
}

// -------- launch ---------------------------------------------------------

extern "C" void kernel_launch(void* const* d_in, const int* in_sizes, int n_in,
                              void* d_out, int out_size, void* d_ws, size_t ws_size,
                              hipStream_t stream) {
    const int*   x     = (const int*)d_in[0];
    const float* emb   = (const float*)d_in[1];
    const float* w_ih  = (const float*)d_in[2];
    const float* w_hh  = (const float*)d_in[3];
    const float* b_ih  = (const float*)d_in[4];
    const float* b_hh  = (const float*)d_in[5];
    const float* fc1_w = (const float*)d_in[6];
    const float* fc1_b = (const float*)d_in[7];
    const float* fc2_w = (const float*)d_in[8];
    const float* fc2_b = (const float*)d_in[9];
    float* out = (float*)d_out;

    char* ws = (char*)d_ws;
    unsigned short* embP  = (unsigned short*)(ws + WS_EMB);
    unsigned short* BhhP  = (unsigned short*)(ws + WS_BHH);
    unsigned short* BihP  = (unsigned short*)(ws + WS_BIH);
    unsigned short* Bfc1P = (unsigned short*)(ws + WS_BFC1);
    float*          biasP = (float*)(ws + WS_BIAS);

    const int embTot = 30000 * EMB_P;              // 3,840,000
    pack_emb_kernel<<<(embTot + 255) / 256, 256, 0, stream>>>(emb, embP, embTot);

    pack_frag_kernel<<<(KT_H * NT_H * 512 + 255) / 256, 256, 0, stream>>>(
        w_hh, BhhP, HID_R, HID_R, NT_H, KT_H * NT_H * 512);
    pack_frag_kernel<<<(KT_E * NT_H * 512 + 255) / 256, 256, 0, stream>>>(
        w_ih, BihP, HID_R, EMB_R, NT_H, KT_E * NT_H * 512);
    pack_frag_kernel<<<(KT_H * NT_F * 512 + 255) / 256, 256, 0, stream>>>(
        fc1_w, Bfc1P, 256, HID_R, NT_F, KT_H * NT_F * 512);
    pack_bias_kernel<<<1, 320, 0, stream>>>(b_ih, b_hh, biasP);

    rnn_fused_kernel<<<BATCH / 64, 128, 0, stream>>>(
        x, embP, BhhP, Bfc1P, biasP, fc1_b, fc2_w, fc2_b, out);
}
